// FeatureSelector_37538014167410
// MI455X (gfx1250) — matrix operand order repaired and checked by reading, not compiled, not run on hardware
//
#include <hip/hip_runtime.h>

typedef __attribute__((ext_vector_type(16))) _Float16     v16h;
typedef __attribute__((ext_vector_type(8)))  float        v8f;
typedef __attribute__((ext_vector_type(4)))  unsigned int v4u;
typedef __attribute__((ext_vector_type(8)))  unsigned int v8u;

#define B_DIM   128
#define L_DIM   16384
#define S_DIM   5000
#define S_PAD   5120          // S padded to multiple of 128 (pad rows = zero mask)
#define KSPLIT1 8             // split of K=L for GEMM1  (2048 per chunk)
#define KSPLIT2 4             // split of K=S_PAD for GEMM2 (1280 per chunk)

// ---------------------------------------------------------------------------
// TDM: issue a 2D f16 tile load (tile_d1 rows x tile_d0 contiguous elems) into
// LDS at lds_byte_off. D# per CDNA5 ISA:
//   group0: [1:0]=count=1, [63:32]=lds_addr, [120:64]=global_addr, [127:126]=2
//   group1: [17:16]=data_size(1 => 2B), dim0/dim1, tile_dim0/1, dim0 stride
// 2D tensors use the short form (VADDR2/VADDR3 = NULL) -> 2 SGPR groups.
// One instruction per wave; completion tracked with TENSORcnt.
// ---------------------------------------------------------------------------
__device__ __forceinline__ void tdm_load_tile_f16(unsigned lds_byte_off,
                                                  const _Float16* gptr,
                                                  unsigned tile_d0, unsigned tile_d1,
                                                  unsigned tensor_d0, unsigned tensor_d1,
                                                  unsigned stride0_elems)
{
    const unsigned long long ga = (unsigned long long)(size_t)gptr;
    v4u g0;
    g0[0] = 1u;                                   // count=1, user descriptor
    g0[1] = lds_byte_off;                         // D#.lds_addr (bytes)
    g0[2] = (unsigned)ga;                         // global_addr[31:0]
    g0[3] = (unsigned)((ga >> 32) & 0x01FFFFFFu)  // global_addr[56:32]
          | (2u << 30);                           // type = 2 ("image")
    v8u g1;
    g1[0] = 1u << 16;                             // data_size=1 (2 bytes); no multicast
    g1[1] = (tensor_d0 & 0xFFFFu) << 16;          // tensor_dim0[15:0] (bits 63:48)
    g1[2] = (tensor_d0 >> 16) | ((tensor_d1 & 0xFFFFu) << 16);
    g1[3] = (tensor_d1 >> 16) | (tile_d0 << 16);  // tensor_dim1 hi | tile_dim0
    g1[4] = tile_d1 & 0xFFFFu;                    // tile_dim1 (tile_dim2 = 0)
    g1[5] = stride0_elems;                        // tensor_dim0_stride[31:0]
    g1[6] = 0u;                                   // stride0[47:32]=0, stride1 lo=0
    g1[7] = 0u;                                   // stride1 hi (unused for 2D)
    asm volatile("tensor_load_to_lds %0, %1" :: "s"(g0), "s"(g1) : "memory");
}

// ---------------------------------------------------------------------------
// Kernel 1: exact top-k(mask_size) per noise row via 4-pass radix select on
// sign-flipped float bits. Emits dense f16 mask row (1.0 / 0.0). Rows >= S_DIM
// are zero padding.
// ---------------------------------------------------------------------------
__global__ __launch_bounds__(256)
void topk_mask_kernel(const float* __restrict__ noise,
                      const int*   __restrict__ kptr,
                      _Float16*    __restrict__ mask)
{
    const int s   = blockIdx.x;
    const int tid = threadIdx.x;
    _Float16* mrow = mask + (size_t)s * L_DIM;

    if (s >= S_DIM) {                       // zero the padding rows
        for (int i = tid; i < L_DIM; i += 256) mrow[i] = (_Float16)0.f;
        return;
    }
    const float* nrow = noise + (size_t)s * L_DIM;

    __shared__ unsigned int hist[256];
    __shared__ unsigned int sh[3];          // [0]=prefix  [1]=still-needed  [2]=tie ctr
    if (tid == 0) { sh[0] = 0u; sh[1] = (unsigned int)kptr[0]; sh[2] = 0u; }

    unsigned int prefixMask = 0u;
    for (int pass = 0; pass < 4; ++pass) {
        const int shift = 24 - 8 * pass;
        __syncthreads();                    // prev selection done / sh visible
        hist[tid] = 0u;
        __syncthreads();
        const unsigned int pfx = sh[0];
        for (int i = tid; i < L_DIM; i += 256) {
            unsigned int u   = __float_as_uint(nrow[i]);
            unsigned int key = (u & 0x80000000u) ? ~u : (u | 0x80000000u);
            if ((key & prefixMask) == pfx)
                atomicAdd(&hist[(key >> shift) & 255u], 1u);
        }
        __syncthreads();
        if (tid == 0) {                     // descending scan: pick bucket
            unsigned int need = sh[1], accum = 0u;
            for (int b = 255; b >= 0; --b) {
                const unsigned int h = hist[b];
                if (accum + h >= need) {
                    sh[0] = pfx | ((unsigned int)b << shift);
                    sh[1] = need - accum;
                    break;
                }
                accum += h;
            }
        }
        prefixMask |= (0xFFu << shift);
    }
    __syncthreads();
    const unsigned int T   = sh[0];         // exact 32-bit threshold key
    const unsigned int rem = sh[1];         // how many == T to take
    for (int i = tid; i < L_DIM; i += 256) {
        unsigned int u   = __float_as_uint(nrow[i]);
        unsigned int key = (u & 0x80000000u) ? ~u : (u | 0x80000000u);
        _Float16 v = (_Float16)0.f;
        if (key > T)                              v = (_Float16)1.f;
        else if (key == T && atomicAdd(&sh[2], 1u) < rem) v = (_Float16)1.f;
        mrow[i] = v;
    }
}

// ---------------------------------------------------------------------------
// Kernel 2: convert saliency a (f32) -> f16 for WMMA A operand
// ---------------------------------------------------------------------------
__global__ __launch_bounds__(256)
void cvt_a_kernel(const float* __restrict__ in, _Float16* __restrict__ out)
{
    const size_t i = (size_t)blockIdx.x * 256 + threadIdx.x;
    out[i] = (_Float16)in[i];
}

// ---------------------------------------------------------------------------
// GEMM1: part1[kc][b][s] = sum_{l in chunk kc} aF16[b][l] * mask[s][l]
// 8 waves; WG computes a 128(M=B) x 128(N=S) tile. Mask row-major == B-operand
// column-contiguous-in-K layout (contraction = l): TDM packs each 128x32 tile
// contiguously into LDS. Double-buffered: wave 0 streams tile i+1 via the
// Tensor Data Mover while all waves compute tile i.
// ---------------------------------------------------------------------------
__global__ __launch_bounds__(256)
void gemm1_kernel(const _Float16* __restrict__ A,
                  const _Float16* __restrict__ Mh,
                  float*          __restrict__ part)
{
    __shared__ __attribute__((aligned(32))) _Float16 lA[2][128 * 32]; // [b][k]
    __shared__ __attribute__((aligned(32))) _Float16 lB[2][128 * 32]; // [s][k]
    const int n0   = blockIdx.x * 128;                 // S tile origin
    const int kc   = blockIdx.y;
    const int kBeg = kc * (L_DIM / KSPLIT1);
    const int kEnd = kBeg + (L_DIM / KSPLIT1);
    const int tid  = threadIdx.x;
    const int wave = tid >> 5, lane = tid & 31;
    const int m0   = wave * 16;
    const int arow = m0 + (lane & 15);
    const int koff = (lane < 16) ? 0 : 16;             // A frag K-half select
    const int kb   = (lane < 16) ? 0 : 16;             // B frag K-half select

    // low 32 bits of a generic LDS pointer == LDS byte offset (ISA 10.2)
    const unsigned ldsA0 = (unsigned)(size_t)&lA[0][0];
    const unsigned ldsA1 = (unsigned)(size_t)&lA[1][0];
    const unsigned ldsB0 = (unsigned)(size_t)&lB[0][0];
    const unsigned ldsB1 = (unsigned)(size_t)&lB[1][0];

    if (wave == 0) {                                   // prologue: tile 0
        tdm_load_tile_f16(ldsA0, &A[kBeg], 32, 128, L_DIM, B_DIM, L_DIM);
        tdm_load_tile_f16(ldsB0, &Mh[(size_t)n0 * L_DIM + kBeg],
                          32, 128, L_DIM, S_PAD, L_DIM);
    }

    v8f acc[8] = {};
    int p = 0;
    for (int k0 = kBeg; k0 < kEnd; k0 += 32) {
        if (wave == 0) __builtin_amdgcn_s_wait_tensorcnt(0); // tile p landed
        __syncthreads();                                     // publish buffer p
        if (wave == 0 && k0 + 32 < kEnd) {                   // stream tile p^1
            tdm_load_tile_f16(p ? ldsA0 : ldsA1, &A[k0 + 32],
                              32, 128, L_DIM, B_DIM, L_DIM);
            tdm_load_tile_f16(p ? ldsB0 : ldsB1,
                              &Mh[(size_t)n0 * L_DIM + k0 + 32],
                              32, 128, L_DIM, S_PAD, L_DIM);
        }
        const _Float16* lAp = lA[p];
        const _Float16* lBp = lB[p];
        v16h af;                                       // A 16x32: 2x16B chunks
        *((uint4*)&af)     = *(const uint4*)&lAp[arow * 32 + koff];
        *((uint4*)&af + 1) = *(const uint4*)&lAp[arow * 32 + koff + 8];
        v16h bf[8];
#pragma unroll
        for (int nt = 0; nt < 8; ++nt)                 // batch the ds loads
            bf[nt] = *(const v16h*)&lBp[(nt * 16 + (lane & 15)) * 32 + kb];
#pragma unroll
        for (int nt = 0; nt < 8; ++nt)
            acc[nt] = __builtin_amdgcn_wmma_f32_16x16x32_f16(
                false, af, false, bf[nt], (short)0, acc[nt], false, false);
        p ^= 1;
    }
    // C/D layout: VGPR r: lanes0-15 -> M=m0+r, lanes16-31 -> M=m0+8+r
    const int mbase = m0 + ((lane < 16) ? 0 : 8);
    const int ncol  = lane & 15;
    float* outp = part + (size_t)kc * B_DIM * S_PAD;
#pragma unroll
    for (int nt = 0; nt < 8; ++nt)
#pragma unroll
        for (int r = 0; r < 8; ++r)
            outp[(size_t)(mbase + r) * S_PAD + n0 + nt * 16 + ncol] = acc[nt][r];
}

// ---------------------------------------------------------------------------
// Kernel 4: per-batch-row softmax stats. Sums the GEMM1 partials in fixed
// order, row-max over valid S, z=exp(am-max) stored f16 (pad rows = 0),
// 1/sum stored for the finalize kernel. Deterministic tree reductions.
// ---------------------------------------------------------------------------
__global__ __launch_bounds__(256)
void softmax_kernel(const float* __restrict__ part,
                    _Float16*    __restrict__ z,
                    float*       __restrict__ invsum)
{
    const int b = blockIdx.x, tid = threadIdx.x;
    __shared__ float am[S_PAD];
    __shared__ float red[256];

    float mx = -3.4e38f;
    for (int i = tid; i < S_PAD; i += 256) {
        float v = 0.f;
#pragma unroll
        for (int c = 0; c < KSPLIT1; ++c)
            v += part[((size_t)c * B_DIM + b) * S_PAD + i];
        am[i] = v;
        if (i < S_DIM && v > mx) mx = v;
    }
    red[tid] = mx; __syncthreads();
    for (int off = 128; off > 0; off >>= 1) {
        if (tid < off) red[tid] = fmaxf(red[tid], red[tid + off]);
        __syncthreads();
    }
    mx = red[0]; __syncthreads();

    float sum = 0.f;
    for (int i = tid; i < S_PAD; i += 256) {
        const float e = (i < S_DIM) ? __expf(am[i] - mx) : 0.f;
        z[(size_t)b * S_PAD + i] = (_Float16)e;
        sum += e;
    }
    red[tid] = sum; __syncthreads();
    for (int off = 128; off > 0; off >>= 1) {
        if (tid < off) red[tid] += red[tid + off];
        __syncthreads();
    }
    if (tid == 0) invsum[b] = 1.f / red[0];
}

// ---------------------------------------------------------------------------
// GEMM2: part2[kc][b][l] = sum_{s in chunk kc} z[b][s] * mask[s][l]
// Contraction = s, so the B operand needs mask columns contiguous in K:
// transpose each 32x128 mask tile into LDS during the fill (TDM cannot
// transpose 16-bit data, so this stays a manual VMEM->LDS fill).
// ---------------------------------------------------------------------------
__global__ __launch_bounds__(256)
void gemm2_kernel(const _Float16* __restrict__ Z,
                  const _Float16* __restrict__ Mh,
                  float*          __restrict__ part)
{
    __shared__ __attribute__((aligned(32))) _Float16 lA[128 * 32]; // [b][k]
    __shared__ __attribute__((aligned(32))) _Float16 lB[128 * 32]; // [l][k] (transposed)
    const int n0   = blockIdx.x * 128;                 // L tile origin
    const int kc   = blockIdx.y;
    const int kBeg = kc * (S_PAD / KSPLIT2);
    const int kEnd = kBeg + (S_PAD / KSPLIT2);
    const int tid  = threadIdx.x;
    const int wave = tid >> 5, lane = tid & 31;
    const int m0   = wave * 16;
    const int arow = m0 + (lane & 15);
    const int koff = (lane < 16) ? 0 : 16;
    const int kb   = (lane < 16) ? 0 : 16;

    v8f acc[8] = {};
    for (int k0 = kBeg; k0 < kEnd; k0 += 32) {
        __syncthreads();
        for (int i = tid; i < 512; i += 256) {         // A tile [128][32]
            const int row = i >> 2, c = (i & 3) << 3;
            *(uint4*)&lA[row * 32 + c] =
                *(const uint4*)&Z[(size_t)row * S_PAD + k0 + c];
        }
        for (int i = tid; i < 512; i += 256) {         // B tile transpose fill
            const int krow = i >> 4;                   // 0..31 (s)
            const int nblk = (i & 15) << 3;            // 0..120 (l, by 8)
            uint4 raw = *(const uint4*)&Mh[(size_t)(k0 + krow) * L_DIM + n0 + nblk];
            const _Float16* hv = (const _Float16*)&raw;
#pragma unroll
            for (int j = 0; j < 8; ++j)
                lB[(nblk + j) * 32 + krow] = hv[j];
        }
        if (k0 + 32 < kEnd) {                          // gfx1250 global_prefetch
            const int krow = tid & 31;
            __builtin_prefetch(
                &Mh[(size_t)(k0 + 32 + krow) * L_DIM + n0 + ((tid >> 5) << 4)], 0, 3);
        }
        __syncthreads();
        v16h af;
        *((uint4*)&af)     = *(const uint4*)&lA[arow * 32 + koff];
        *((uint4*)&af + 1) = *(const uint4*)&lA[arow * 32 + koff + 8];
        v16h bf[8];
#pragma unroll
        for (int nt = 0; nt < 8; ++nt)
            bf[nt] = *(const v16h*)&lB[(nt * 16 + (lane & 15)) * 32 + kb];
#pragma unroll
        for (int nt = 0; nt < 8; ++nt)
            acc[nt] = __builtin_amdgcn_wmma_f32_16x16x32_f16(
                false, af, false, bf[nt], (short)0, acc[nt], false, false);
    }
    const int mbase = m0 + ((lane < 16) ? 0 : 8);
    const int ncol  = lane & 15;
    float* outp = part + (size_t)kc * B_DIM * L_DIM;
#pragma unroll
    for (int nt = 0; nt < 8; ++nt)
#pragma unroll
        for (int r = 0; r < 8; ++r)
            outp[(size_t)(mbase + r) * L_DIM + n0 + nt * 16 + ncol] = acc[nt][r];
}

// ---------------------------------------------------------------------------
// Kernel 6: out[b][l] = (sum_kc part2[kc][b][l]) * invsum[b]   (fixed order)
// ---------------------------------------------------------------------------
__global__ __launch_bounds__(256)
void finalize_kernel(const float* __restrict__ part,
                     const float* __restrict__ invsum,
                     float*       __restrict__ out)
{
    const size_t i = (size_t)blockIdx.x * 256 + threadIdx.x;  // over B*L
    const int b = (int)(i >> 14);                             // / L_DIM
    float v = 0.f;
#pragma unroll
    for (int c = 0; c < KSPLIT2; ++c)
        v += part[(size_t)c * B_DIM * L_DIM + i];
    out[i] = v * invsum[b];
}

// ---------------------------------------------------------------------------
extern "C" void kernel_launch(void* const* d_in, const int* in_sizes, int n_in,
                              void* d_out, int out_size, void* d_ws, size_t ws_size,
                              hipStream_t stream)
{
    (void)in_sizes; (void)n_in; (void)out_size; (void)ws_size;
    const float* a_in  = (const float*)d_in[0];   // [128,1,128,128] f32
    const float* noise = (const float*)d_in[1];   // [5000,16384]    f32
    const int*   kptr  = (const int*)  d_in[2];   // scalar mask_size
    float*       out   = (float*)d_out;           // [128,1,128,128] f32

    char*  ws  = (char*)d_ws;
    size_t off = 0;
    auto alloc = [&](size_t bytes) -> void* {
        void* p = ws + off;
        off = (off + bytes + 255) & ~(size_t)255;
        return p;
    };
    _Float16* mask = (_Float16*)alloc((size_t)S_PAD * L_DIM * 2);          // 168 MB
    _Float16* aF16 = (_Float16*)alloc((size_t)B_DIM * L_DIM * 2);          //   4 MB
    float*    p1   = (float*)   alloc((size_t)KSPLIT1 * B_DIM * S_PAD * 4);//  21 MB
    _Float16* zF16 = (_Float16*)alloc((size_t)B_DIM * S_PAD * 2);          // 1.3 MB
    float*    isum = (float*)   alloc((size_t)B_DIM * 4);
    float*    p2   = (float*)   alloc((size_t)KSPLIT2 * B_DIM * L_DIM * 4);//  34 MB

    topk_mask_kernel<<<S_PAD, 256, 0, stream>>>(noise, kptr, mask);
    cvt_a_kernel<<<(B_DIM * L_DIM) / 256, 256, 0, stream>>>(a_in, aF16);
    gemm1_kernel<<<dim3(S_PAD / 128, KSPLIT1), 256, 0, stream>>>(aF16, mask, p1);
    softmax_kernel<<<B_DIM, 256, 0, stream>>>(p1, zF16, isum);
    gemm2_kernel<<<dim3(L_DIM / 128, KSPLIT2), 256, 0, stream>>>(zF16, mask, p2);
    finalize_kernel<<<(B_DIM * L_DIM) / 256, 256, 0, stream>>>(p2, isum, out);
}
